// Recurrent_17480516895033
// MI455X (gfx1250) — compile-verified, matrix operand order checked by reading, not run on hardware
//
#include <hip/hip_runtime.h>
#include <hip/hip_bf16.h>

#define Hd   1024
#define Bd   64
#define Td   256
#define Ld   4
#define Od   1024
#define LNE  1e-5f
#define NWG  32          // workgroups in persistent layer kernel

typedef __bf16 bf16;
typedef __attribute__((ext_vector_type(16))) __bf16 v16bf;
typedef __attribute__((ext_vector_type(8)))  __bf16 bf16x8;
typedef __attribute__((ext_vector_type(8)))  float  v8f;
typedef __attribute__((ext_vector_type(4)))  unsigned int u32x4;
typedef __attribute__((ext_vector_type(8)))  unsigned int u32x8;

__device__ __forceinline__ bf16 f2bf(float f) {
  union { float f; unsigned int u; } x; x.f = f;
  unsigned int r = (x.u + 0x7FFFu + ((x.u >> 16) & 1u)) >> 16;
  union { unsigned short s; bf16 b; } y; y.s = (unsigned short)r;
  return y.b;
}

__device__ __forceinline__ v16bf combine16(bf16x8 lo, bf16x8 hi) {
  v16bf r;
#pragma unroll
  for (int i = 0; i < 8; ++i) { r[i] = lo[i]; r[i + 8] = hi[i]; }
  return r;
}

// 16x32 bf16 fragment load (generic pointer version, used for LDS tiles).
__device__ __forceinline__ v16bf load_frag(const bf16* base, int ld) {
  int lane = threadIdx.x & 31;
  int row  = lane & 15;
  int half = lane >> 4;
  const bf16* p = base + row * ld + half * 8;
  return combine16(*(const bf16x8*)(p), *(const bf16x8*)(p + 16));
}

// ---- CDNA5 async global->LDS copy (ASYNCcnt path) ----
__device__ __forceinline__ void async_g2l_b128(unsigned lds_off, const void* gaddr) {
  asm volatile("global_load_async_to_lds_b128 %0, %1, off"
               :: "v"(lds_off), "v"((unsigned long long)(size_t)gaddr)
               : "memory");
}
__device__ __forceinline__ void wait_async0() {
  asm volatile("s_wait_asynccnt 0x0" ::: "memory");
}

// ---- CDNA5 Tensor Data Mover: 2-group descriptor form (VADDR2/3 NULL) ----
__device__ __forceinline__ void tdm_load_to_lds(u32x4 g0, u32x8 g1) {
  asm volatile("tensor_load_to_lds %0, %1" :: "s"(g0), "s"(g1) : "memory");
}
__device__ __forceinline__ void wait_tensor0() {
  __builtin_amdgcn_s_wait_tensorcnt(0);
}

// ---- agent-scope grid barrier (monotonic counter per barrier slot) ----
__device__ __forceinline__ void grid_barrier(int* ctr, int expected) {
  __syncthreads();
  if (threadIdx.x == 0) {
    __hip_atomic_fetch_add(ctr, 1, __ATOMIC_RELEASE, __HIP_MEMORY_SCOPE_AGENT);
    while (__hip_atomic_load(ctr, __ATOMIC_ACQUIRE, __HIP_MEMORY_SCOPE_AGENT) < expected) {
      __builtin_amdgcn_s_sleep(1);
    }
  }
  __syncthreads();
}

// ---------------- helpers ----------------
__global__ void f2bf_kernel(const float* __restrict__ in, bf16* __restrict__ out, int n) {
  int i = blockIdx.x * blockDim.x + threadIdx.x;
  if (i < n) out[i] = f2bf(in[i]);
}
__global__ void zero_bf_kernel(bf16* __restrict__ p, int n) {
  int i = blockIdx.x * blockDim.x + threadIdx.x;
  if (i < n) p[i] = f2bf(0.0f);
}
__global__ void zero_i32_kernel(int* __restrict__ p, int n) {
  int i = blockIdx.x * blockDim.x + threadIdx.x;
  if (i < n) p[i] = 0;
}

// ---------------- big GEMM: out = A[M,K](bf16) * W[N,K]^T(bf16) + bias ----------------
// Fixed dims M=16384, N=K=1024. Wave tile 16x128 (8 accs). Ping-pong 2-stage
// pipeline (k += 64) with per-lane base pointers + immediate offsets: no register
// copies, no spills (launch_bounds gives the allocator a 1-wave/SIMD floor).
// NOTE: the steady-state prefetch overreads one 32-wide K-chunk past K on the last
// iteration; all GEMM operands sit in contiguous workspace carves so this is safe.
__global__ __launch_bounds__(256, 1)
void gemm_bf16_kernel(const bf16* __restrict__ A,
                      const bf16* __restrict__ W,
                      const float* __restrict__ bias,
                      float* __restrict__ out,
                      int remapTB) {
  constexpr int M = Bd * Td;   // 16384
  constexpr int N = Hd;        // 1024
  constexpr int K = Hd;        // 1024
  int wid = blockIdx.x * 8 + (threadIdx.x >> 5);
  constexpr int nstrips = N / 128;       // 8
  int mt = wid / nstrips;
  int ns = wid - mt * nstrips;
  if (mt * 16 >= M) return;
  int m0 = mt * 16, n0 = ns * 128;
  int lane = threadIdx.x & 31;
  int cn   = lane & 15;
  int half = lane >> 4;
  int mb   = half * 8;

  v8f acc[8];
#pragma unroll
  for (int j = 0; j < 8; ++j) {
    float bv = bias[n0 + j * 16 + cn];
#pragma unroll
    for (int r = 0; r < 8; ++r) acc[j][r] = bv;
  }

  // Per-lane base pointers (row = cn for both A and W fragments).
  const bf16* aRow = A + (size_t)(m0 + cn) * K + half * 8;
  const bf16* wRow = W + (size_t)(n0 + cn) * K + half * 8;

  auto ldA = [&](int k) {
    return combine16(*(const bf16x8*)(aRow + k), *(const bf16x8*)(aRow + k + 16));
  };
  auto ldB = [&](int j, int k) {
    const bf16* p = wRow + j * 16 * K + k;   // j*32768B -> immediate offset
    return combine16(*(const bf16x8*)(p), *(const bf16x8*)(p + 16));
  };

  v16bf a[2], b[2][8];
  a[0] = ldA(0);
#pragma unroll
  for (int j = 0; j < 8; ++j) b[0][j] = ldB(j, 0);

  for (int k = 0; k < K; k += 64) {
    a[1] = ldA(k + 32);
#pragma unroll
    for (int j = 0; j < 8; ++j) b[1][j] = ldB(j, k + 32);
#pragma unroll
    for (int j = 0; j < 8; ++j)
      acc[j] = __builtin_amdgcn_wmma_f32_16x16x32_bf16(false, a[0], false, b[0][j],
                                                       (short)0, acc[j], false, false);
    a[0] = ldA(k + 64);                       // last iter overreads into adjacent carve
#pragma unroll
    for (int j = 0; j < 8; ++j) b[0][j] = ldB(j, k + 64);
#pragma unroll
    for (int j = 0; j < 8; ++j)
      acc[j] = __builtin_amdgcn_wmma_f32_16x16x32_bf16(false, a[1], false, b[1][j],
                                                       (short)0, acc[j], false, false);
  }

#pragma unroll
  for (int j = 0; j < 8; ++j) {
    int n = n0 + j * 16 + cn;
#pragma unroll
    for (int r = 0; r < 8; ++r) {
      int m = m0 + mb + r;
      size_t idx;
      if (remapTB) { int b2 = m / Td; int t = m - b2 * Td; idx = ((size_t)t * Bd + b2) * N + n; }
      else         { idx = (size_t)m * N + n; }
      out[idx] = acc[j][r];
    }
  }
}

// ---------------- persistent per-layer recurrent kernel ----------------
// 32 WGs x 256 thr. WG wg owns output columns [wg*32, wg*32+32).
// LDS: Wh strip (32x1024 bf16, pinned once via async-to-LDS) + h (64x1024 bf16,
// refreshed per step via one TDM tensor_load_to_lds) + LN partials.
__global__ __launch_bounds__(256, 1)
void layer_rnn_kernel(const bf16* __restrict__ Wh_l,   // [1024,1024]
                      const float* __restrict__ xproj, // [T,64,1024]
                      const float* __restrict__ g,     // [1024]
                      const float* __restrict__ be,    // [1024]
                      bf16* __restrict__ hglob,        // [64,1024] pre-zeroed
                      bf16* __restrict__ xout_bf,      // [B,T,H]
                      float* __restrict__ xout_f32,    // nullable
                      float* __restrict__ gsum,        // [T*64] zeroed
                      float* __restrict__ gsq,         // [T*64] zeroed
                      int* __restrict__ ctrs) {        // [2*T] zeroed
  extern __shared__ char smem[];
  bf16*  WhS = (bf16*)smem;                       // 64 KB
  bf16*  hS  = (bf16*)(smem + 64 * 1024);         // 128 KB
  float* rs  = (float*)(smem + 192 * 1024);       // 64 floats
  float* rq  = rs + 64;

  const int wg   = blockIdx.x;
  const int tid  = threadIdx.x;
  const int wave = tid >> 5;
  const int lane = tid & 31;
  const int mt   = wave >> 1;
  const int nsub = wave & 1;
  const int m0   = mt * 16;
  const int n0   = wg * 32 + nsub * 16;
  const int cn   = lane & 15;
  const int mb   = (lane >> 4) * 8;
  const int ncol = n0 + cn;

  const unsigned whsBase = (unsigned)(size_t)(void*)WhS;
  const unsigned hsBase  = (unsigned)(size_t)(void*)hS;

  // TDM descriptor for the per-step h copy (constant across steps):
  // 1-D tile of 16384 x 8-byte elements = 128 KB, global hglob -> LDS hS.
  unsigned long long ga = (unsigned long long)(size_t)hglob;
  u32x4 g0;
  g0[0] = 1u;                                               // count=1
  g0[1] = hsBase;                                           // lds_addr
  g0[2] = (unsigned)(ga & 0xFFFFFFFFu);                     // global_addr[31:0]
  g0[3] = (unsigned)((ga >> 32) & 0x1FFFFFFu) | (2u << 30); // addr[56:32] | type=2
  u32x8 g1;
  g1[0] = (3u << 16);                                       // data_size = 8 bytes
  g1[1] = (16384u & 0xFFFFu) << 16;                         // tensor_dim0[15:0]
  g1[2] = (1u << 16);                                       // dim0 hi | tensor_dim1=1
  g1[3] = (16384u & 0xFFFFu) << 16;                         // tile_dim0 = 16384
  g1[4] = 0u;
  g1[5] = 16384u;                                           // tensor_dim0_stride
  g1[6] = 0u;
  g1[7] = 0u;

  // Pin this WG's 32-row strip of Wh into LDS once (async-to-LDS, 64 KB).
  {
    const char* src = (const char*)(Wh_l + (size_t)wg * 32 * Hd);
#pragma unroll
    for (int it = 0; it < 16; ++it) {
      unsigned off = (unsigned)((it * 256 + tid) * 16);
      async_g2l_b128(whsBase + off, src + off);
    }
    wait_async0();
  }
  __syncthreads();

  const float gcol = g[ncol];
  const float bcol = be[ncol];

  for (int t = 0; t < Td; ++t) {
    // ---- stage h (128 KB) into LDS with one TDM op from wave 0 ----
    if (wave == 0) tdm_load_to_lds(g0, g1);
    if (tid < 64) { rs[tid] = 0.f; rq[tid] = 0.f; }
    if (wave == 0) wait_tensor0();
    __syncthreads();

    // ---- z tile = xproj_t + h * Wh^T (WMMA from LDS, pipelined) ----
    const float* xpt = xproj + (size_t)t * Bd * Hd;
    v8f acc;
#pragma unroll
    for (int r = 0; r < 8; ++r)
      acc[r] = xpt[(size_t)(m0 + mb + r) * Hd + ncol];

    {
      const bf16* ha = hS + (size_t)m0 * Hd;
      const bf16* wb = WhS + (size_t)(nsub * 16) * Hd;
      v16bf a0 = load_frag(ha, Hd);
      v16bf w0 = load_frag(wb, Hd);
#pragma unroll
      for (int k = 32; k < Hd; k += 32) {
        v16bf a1 = load_frag(ha + k, Hd);
        v16bf w1 = load_frag(wb + k, Hd);
        acc = __builtin_amdgcn_wmma_f32_16x16x32_bf16(false, a0, false, w0,
                                                      (short)0, acc, false, false);
        a0 = a1; w0 = w1;
      }
      acc = __builtin_amdgcn_wmma_f32_16x16x32_bf16(false, a0, false, w0,
                                                    (short)0, acc, false, false);
    }

    // ---- LN partial sums: LDS ds_add_f32, then one global atomic per row ----
#pragma unroll
    for (int r = 0; r < 8; ++r) {
      int row = m0 + mb + r;
      atomicAdd(&rs[row], acc[r]);
      atomicAdd(&rq[row], acc[r] * acc[r]);
    }
    __syncthreads();
    if (tid < 64) {
      __hip_atomic_fetch_add(&gsum[t * 64 + tid], rs[tid],
                             __ATOMIC_RELAXED, __HIP_MEMORY_SCOPE_AGENT);
      __hip_atomic_fetch_add(&gsq[t * 64 + tid], rq[tid],
                             __ATOMIC_RELAXED, __HIP_MEMORY_SCOPE_AGENT);
    }
    grid_barrier(&ctrs[2 * t], NWG);

    // ---- normalize + tanh in registers, write h / layer output ----
#pragma unroll
    for (int r = 0; r < 8; ++r) {
      int row = m0 + mb + r;
      float mu  = gsum[t * 64 + row] * (1.0f / Hd);
      float var = gsq[t * 64 + row] * (1.0f / Hd) - mu * mu;
      float inv = rsqrtf(var + LNE);
      float val = tanhf((acc[r] - mu) * inv * gcol + bcol);
      bf16 bv = f2bf(val);
      hglob[(size_t)row * Hd + ncol] = bv;
      size_t ob = (size_t)row * Td * Hd + (size_t)t * Hd + ncol;
      xout_bf[ob] = bv;
      if (xout_f32) xout_f32[ob] = val;
    }
    grid_barrier(&ctrs[2 * t + 1], NWG);
  }
}

extern "C" void kernel_launch(void* const* d_in, const int* in_sizes, int n_in,
                              void* d_out, int out_size, void* d_ws, size_t ws_size,
                              hipStream_t stream) {
  const float* inp  = (const float*)d_in[0];  // [B,T,H]
  const float* Wx   = (const float*)d_in[1];  // [L,H,H]
  const float* bx   = (const float*)d_in[2];  // [L,H]
  const float* Wh   = (const float*)d_in[3];  // [L,H,H]
  const float* ln_g = (const float*)d_in[4];  // [L,H]
  const float* ln_b = (const float*)d_in[5];  // [L,H]
  const float* Wy   = (const float*)d_in[6];  // [O,H]
  const float* by   = (const float*)d_in[7];  // [O]

  float* out_x = (float*)d_out;                 // [B,T,H] fp32
  float* out_y = out_x + (size_t)Bd * Td * Hd;  // [B,T,O] fp32

  size_t off = 0;
  auto carve = [&](size_t bytes) {
    void* p = (char*)d_ws + off;
    off += (bytes + 255) & ~(size_t)255;
    return p;
  };
  bf16*  wx_bf = (bf16*) carve((size_t)Ld * Hd * Hd * 2);  // 8 MB
  bf16*  wh_bf = (bf16*) carve((size_t)Ld * Hd * Hd * 2);  // 8 MB
  bf16*  wy_bf = (bf16*) carve((size_t)Od * Hd * 2);       // 2 MB
  bf16*  xcur  = (bf16*) carve((size_t)Bd * Td * Hd * 2);  // 32 MB  [B,T,H]
  float* xproj = (float*)carve((size_t)Td * Bd * Hd * 4);  // 64 MB  [T,B,H]
  bf16*  hb    = (bf16*) carve((size_t)Bd * Hd * 2);       // 128 KB
  size_t statsInts = (size_t)Ld * Td * 64 * 2 + (size_t)Ld * 2 * Td;
  int*   stats = (int*)carve(statsInts * 4);
  float* gsum  = (float*)stats;                        // [L][T*64]
  float* gsq   = gsum + (size_t)Ld * Td * 64;          // [L][T*64]
  int*   ctrs  = (int*)(gsq + (size_t)Ld * Td * 64);   // [L][2*T]

  const int thr = 256;
  zero_i32_kernel<<<(int)((statsInts + thr - 1) / thr), thr, 0, stream>>>(stats, (int)statsInts);

  {
    int n = Ld * Hd * Hd;
    f2bf_kernel<<<(n + thr - 1) / thr, thr, 0, stream>>>(Wx, wx_bf, n);
    f2bf_kernel<<<(n + thr - 1) / thr, thr, 0, stream>>>(Wh, wh_bf, n);
  }
  { int n = Od * Hd;      f2bf_kernel<<<(n + thr - 1) / thr, thr, 0, stream>>>(Wy, wy_bf, n); }
  { int n = Bd * Td * Hd; f2bf_kernel<<<(n + thr - 1) / thr, thr, 0, stream>>>(inp, xcur, n); }

  const int M = Bd * Td;                              // 16384
  const int gemmBlocks = (M / 16) * (Hd / 128) / 8;   // 1024
  const size_t smemBytes = 192 * 1024 + 512;

  for (int l = 0; l < Ld; ++l) {
    gemm_bf16_kernel<<<gemmBlocks, thr, 0, stream>>>(
        xcur, wx_bf + (size_t)l * Hd * Hd, bx + (size_t)l * Hd,
        xproj, /*remapTB=*/1);
    zero_bf_kernel<<<(Bd * Hd + thr - 1) / thr, thr, 0, stream>>>(hb, Bd * Hd);

    layer_rnn_kernel<<<NWG, thr, smemBytes, stream>>>(
        wh_bf + (size_t)l * Hd * Hd, xproj,
        ln_g + (size_t)l * Hd, ln_b + (size_t)l * Hd,
        hb, xcur, (l == Ld - 1) ? out_x : nullptr,
        gsum + (size_t)l * Td * 64, gsq + (size_t)l * Td * 64,
        ctrs + (size_t)l * 2 * Td);
  }

  gemm_bf16_kernel<<<gemmBlocks, thr, 0, stream>>>(
      xcur, wy_bf, by, out_y, /*remapTB=*/0);
}
